// MHA_12403865551605
// MI455X (gfx1250) — matrix operand order repaired and checked by reading, not compiled, not run on hardware
//
#include <hip/hip_runtime.h>

typedef __bf16 bf16;
typedef __attribute__((ext_vector_type(16))) __bf16 v16bf;
typedef __attribute__((ext_vector_type(8)))  float  v8f;

#define D_MODEL  1024
#define N_HEADS  16
#define HEAD_DIM 64
#define BATCH    2
#define SEQ      2048
#define M_TOK    (BATCH*SEQ)     /* 4096 */
#define N_QKV    (3*D_MODEL)     /* 3072 */
#define NEG_BIG  (-1.0e30f)

// ---------------------------------------------------------------------------
// WMMA fragment loaders (contiguous k order k = 16*(lane>>4) + e for both A and B, wave32)
// ---------------------------------------------------------------------------

// 16-bit A matrix 16x32 (MxK): lane&15 = row M; lanes 0-15 hold K=0..15;
// lanes 16-31 hold K=16..31.
static __device__ __forceinline__ v16bf load_a_frag(const bf16* A, int lda, int row, int k0, int lane) {
  const int half = lane >> 4;
  const bf16* p = A + (size_t)row * lda + k0;
  v16bf a;
#pragma unroll
  for (int e = 0; e < 8; ++e) a[e]     = p[16*half + e];
#pragma unroll
  for (int e = 0; e < 8; ++e) a[8 + e] = p[16*half + 8 + e];
  return a;
}

// B fragment 32x16 (KxN) from a matrix stored "N-major" (Bm[n][k], leading dim ldb):
// lane<16: N=lane, K=e ; lane>=16: N=lane-16, K=16+e  => per-lane 32 contiguous bytes.
static __device__ __forceinline__ v16bf load_bT_frag(const bf16* Bm, int ldb, int n_base, int k0, int lane) {
  const bf16* p = Bm + (size_t)(n_base + (lane & 15)) * ldb + k0 + 16*(lane >> 4);
  v16bf b;
#pragma unroll
  for (int e = 0; e < 16; ++e) b[e] = p[e];
  return b;
}

// B fragment from pre-swizzled weights: block (nt, kc) holds 32 lanes x 16 bf16
// contiguous, already in fragment order. One 32B contiguous load per lane.
static __device__ __forceinline__ v16bf load_b_packed(const bf16* Wp, int Kdim, int nt, int kc, int lane) {
  const bf16* p = Wp + (((size_t)nt * (Kdim/32) + kc) * 32 + lane) * 16;
  return *reinterpret_cast<const v16bf*>(p);
}

static __device__ __forceinline__ v8f wmma_bf16(v16bf a, v16bf b, v8f c) {
  return __builtin_amdgcn_wmma_f32_16x16x32_bf16(false, a, false, b, (short)0, c, false, false);
}

static __device__ __forceinline__ v8f vzero() {
  return (v8f){0.f,0.f,0.f,0.f,0.f,0.f,0.f,0.f};
}

// ---------------------------------------------------------------------------
// fp32 -> bf16 conversion (grid-stride)
// ---------------------------------------------------------------------------
__global__ void __launch_bounds__(256) cvt_f32_bf16(const float* __restrict__ in,
                                                    bf16* __restrict__ out, int n) {
  for (int i = blockIdx.x * blockDim.x + threadIdx.x; i < n; i += gridDim.x * blockDim.x)
    out[i] = (bf16)in[i];
}

// ---------------------------------------------------------------------------
// Pack a KxN row-major fp32 weight matrix into bf16 B-fragment order:
// out[((nt*(K/32)+kc)*32 + lane)*16 + e] = W[(kc*32 + 16*(lane>>4) + e)*N + nt*16 + (lane&15)]
// ---------------------------------------------------------------------------
__global__ void __launch_bounds__(256) pack_w_frag(const float* __restrict__ W,
                                                   bf16* __restrict__ out,
                                                   int K, int N) {
  const int total = K * N;
  for (int idx = blockIdx.x * blockDim.x + threadIdx.x; idx < total; idx += gridDim.x * blockDim.x) {
    const int e    = idx & 15;
    const int lane = (idx >> 4) & 31;
    const int t    = idx >> 9;
    const int kc   = t % (K/32);
    const int nt   = t / (K/32);
    const int k = kc*32 + 16*(lane >> 4) + e;
    const int n = nt*16 + (lane & 15);
    out[idx] = (bf16)W[(size_t)k * N + n];
  }
}

// ---------------------------------------------------------------------------
// QKV GEMM: Y = xb(4096x1024) @ Wqkv(1024x3072) + b. One wave = 32x64 tile
// (2 A-frags x 4 B-frags -> 8 WMMAs per K-step). Outputs scattered into
// q,k: [B,H,S,dh] bf16 ; v: TRANSPOSED [B,H,dh,S] bf16 (fragment-friendly PV).
// ---------------------------------------------------------------------------
__global__ void __launch_bounds__(256) qkv_gemm(const bf16* __restrict__ xb,
                                                const bf16* __restrict__ wp,
                                                const float* __restrict__ bias,
                                                bf16* __restrict__ qb,
                                                bf16* __restrict__ kb,
                                                bf16* __restrict__ vt) {
  const int wave = blockIdx.x * 8 + (threadIdx.x >> 5);
  const int lane = threadIdx.x & 31;
  const int lrow = lane & 15;
  const int half = lane >> 4;
  const int NT4 = N_QKV / 64;              // 48
  const int mt2 = wave / NT4, nt4 = wave % NT4;

  v8f acc[2][4];
#pragma unroll
  for (int i = 0; i < 2; ++i)
#pragma unroll
    for (int j = 0; j < 4; ++j) acc[i][j] = vzero();

  const int row0 = mt2*32 + lrow;
  for (int kc = 0; kc < D_MODEL/32; ++kc) {
    __builtin_prefetch((const void*)(xb + (size_t)row0 * D_MODEL + kc*32 + 64), 0, 1);
    v16bf a0 = load_a_frag(xb, D_MODEL, row0,      kc*32, lane);
    v16bf a1 = load_a_frag(xb, D_MODEL, row0 + 16, kc*32, lane);
#pragma unroll
    for (int j = 0; j < 4; ++j) {
      v16bf b = load_b_packed(wp, D_MODEL, nt4*4 + j, kc, lane);
      acc[0][j] = wmma_bf16(a0, b, acc[0][j]);
      acc[1][j] = wmma_bf16(a1, b, acc[1][j]);
    }
  }

  // Epilogue. A 64-wide col group never straddles q/k/v or head boundaries
  // (1024 % 64 == 0), so 'which' and 'h' are uniform over the wave.
#pragma unroll
  for (int j = 0; j < 4; ++j) {
    const int col   = (nt4*4 + j)*16 + lrow;
    const int which = col >> 10;
    const int rem   = col & 1023;
    const int h     = rem >> 6;
    const int d     = rem & 63;
    const float bv  = bias[col];
#pragma unroll
    for (int i = 0; i < 2; ++i)
#pragma unroll
      for (int r = 0; r < 8; ++r) {
        const int m  = mt2*32 + i*16 + r + 8*half;   // global token index
        const int bb = m >> 11;                      // / SEQ
        const int s  = m & (SEQ - 1);
        const bf16 val = (bf16)(acc[i][j][r] + bv);
        if (which == 2)
          vt[((size_t)(bb*N_HEADS + h)*HEAD_DIM + d)*SEQ + s] = val;           // [B,H,dh,S]
        else {
          bf16* dst = (which == 0) ? qb : kb;
          dst[((size_t)(bb*N_HEADS + h)*SEQ + s)*HEAD_DIM + d] = val;          // [B,H,S,dh]
        }
      }
  }
}

// ---------------------------------------------------------------------------
// Flash attention: one wave per (b, h, 32-query-row tile). 32 keys/iteration:
// 8 score WMMAs + 8 PV WMMAs, K/V fragments shared across both query tiles.
// ---------------------------------------------------------------------------
__global__ void __launch_bounds__(256) attn_kernel(const bf16* __restrict__ qg,
                                                   const bf16* __restrict__ kg,
                                                   const bf16* __restrict__ vtg,
                                                   const int*  __restrict__ mask,
                                                   bf16* __restrict__ attnb) {
  __shared__ bf16 lds_p[16][16*32];         // per (wave, qi) 16x32 bf16 prob tile (16 KB)

  const int wave = blockIdx.x * 8 + (threadIdx.x >> 5);
  const int wib  = threadIdx.x >> 5;
  const int lane = threadIdx.x & 31;
  const int lrow = lane & 15;
  const int half = lane >> 4;

  const int qt2 = wave & 63;                // S/32 = 64 query tiles
  const int h   = (wave >> 6) & (N_HEADS - 1);
  const int b   = wave >> 10;

  const size_t head_off = (size_t)(b*N_HEADS + h) * SEQ * HEAD_DIM;
  const bf16* Q  = qg  + head_off;
  const bf16* K  = kg  + head_off;
  const bf16* Vt = vtg + head_off;          // [dh][S]
  const int* mrow = mask + b*SEQ;

  const int q0 = qt2 * 32;
  v16bf aq[2][2];
#pragma unroll
  for (int qi = 0; qi < 2; ++qi)
#pragma unroll
    for (int c = 0; c < 2; ++c)
      aq[qi][c] = load_a_frag(Q, HEAD_DIM, q0 + qi*16 + lrow, c*32, lane);

  v8f acc[2][4];
#pragma unroll
  for (int qi = 0; qi < 2; ++qi)
#pragma unroll
    for (int t = 0; t < 4; ++t) acc[qi][t] = vzero();
  float run_m[2][8], run_s[2][8];
#pragma unroll
  for (int qi = 0; qi < 2; ++qi)
#pragma unroll
    for (int r = 0; r < 8; ++r) { run_m[qi][r] = NEG_BIG; run_s[qi][r] = 0.f; }

  for (int kt = 0; kt < SEQ/32; ++kt) {
    const int key0 = kt * 32;

    // K^T fragments (shared by both query tiles): per-lane contiguous 32B loads
    const v16bf kf00 = load_bT_frag(K, HEAD_DIM, key0,      0,  lane);
    const v16bf kf01 = load_bT_frag(K, HEAD_DIM, key0,      32, lane);
    const v16bf kf10 = load_bT_frag(K, HEAD_DIM, key0 + 16, 0,  lane);
    const v16bf kf11 = load_bT_frag(K, HEAD_DIM, key0 + 16, 32, lane);

    // V fragments from transposed layout (shared): contiguous 32B loads
    v16bf vf[4];
#pragma unroll
    for (int t = 0; t < 4; ++t) vf[t] = load_bT_frag(Vt, SEQ, t*16, key0, lane);

    const float mb0 = (mrow[key0 + lrow]      != 0) ? 0.f : NEG_BIG;
    const float mb1 = (mrow[key0 + 16 + lrow] != 0) ? 0.f : NEG_BIG;

#pragma unroll
    for (int qi = 0; qi < 2; ++qi) {
      v8f s0 = wmma_bf16(aq[qi][0], kf00, vzero());
      s0     = wmma_bf16(aq[qi][1], kf01, s0);
      v8f s1 = wmma_bf16(aq[qi][0], kf10, vzero());
      s1     = wmma_bf16(aq[qi][1], kf11, s1);

      const int slot = wib*2 + qi;
#pragma unroll
      for (int r = 0; r < 8; ++r) {
        float f0 = s0[r] * 0.125f + mb0;    // 1/sqrt(64)
        float f1 = s1[r] * 0.125f + mb1;

        float tmax = fmaxf(f0, f1);
#pragma unroll
        for (int off = 8; off > 0; off >>= 1) tmax = fmaxf(tmax, __shfl_xor(tmax, off, 16));

        const float nm   = fmaxf(run_m[qi][r], tmax);
        const float corr = __expf(run_m[qi][r] - nm);
        run_m[qi][r] = nm;

        const float p0 = __expf(f0 - nm);
        const float p1 = __expf(f1 - nm);
        float ps = p0 + p1;
#pragma unroll
        for (int off = 8; off > 0; off >>= 1) ps += __shfl_xor(ps, off, 16);
        run_s[qi][r] = run_s[qi][r] * corr + ps;

        // C-fragment layout -> LDS (same-wave DS ops are in-order; no barrier)
        lds_p[slot][(r + 8*half)*32 + lrow]      = (bf16)p0;
        lds_p[slot][(r + 8*half)*32 + 16 + lrow] = (bf16)p1;

        acc[qi][0][r] *= corr; acc[qi][1][r] *= corr;
        acc[qi][2][r] *= corr; acc[qi][3][r] *= corr;
      }

      // re-read probs as A fragment (16x32) and do PV with shared V fragments
      const v16bf pa = load_a_frag(&lds_p[slot][0], 32, lrow, 0, lane);
#pragma unroll
      for (int t = 0; t < 4; ++t) acc[qi][t] = wmma_bf16(pa, vf[t], acc[qi][t]);
    }
  }

  // normalize and store to [B, S, H*dh] bf16 (ready to be proj-GEMM's A)
#pragma unroll
  for (int qi = 0; qi < 2; ++qi)
#pragma unroll
    for (int t = 0; t < 4; ++t)
#pragma unroll
      for (int r = 0; r < 8; ++r) {
        const int m = q0 + qi*16 + r + 8*half;
        attnb[(size_t)(b*SEQ + m) * D_MODEL + h*HEAD_DIM + t*16 + lrow] =
            (bf16)(acc[qi][t][r] / run_s[qi][r]);
      }
}

// ---------------------------------------------------------------------------
// Output projection: out = attn(4096x1024) @ Wproj(1024x1024) + b (fp32 out).
// One wave = 32x64 tile, packed weights.
// ---------------------------------------------------------------------------
__global__ void __launch_bounds__(256) proj_gemm(const bf16* __restrict__ A,
                                                 const bf16* __restrict__ wp,
                                                 const float* __restrict__ bias,
                                                 float* __restrict__ out) {
  const int wave = blockIdx.x * 8 + (threadIdx.x >> 5);
  const int lane = threadIdx.x & 31;
  const int lrow = lane & 15;
  const int half = lane >> 4;
  const int NT4 = D_MODEL / 64;             // 16
  const int mt2 = wave / NT4, nt4 = wave % NT4;

  v8f acc[2][4];
#pragma unroll
  for (int i = 0; i < 2; ++i)
#pragma unroll
    for (int j = 0; j < 4; ++j) acc[i][j] = vzero();

  const int row0 = mt2*32 + lrow;
  for (int kc = 0; kc < D_MODEL/32; ++kc) {
    __builtin_prefetch((const void*)(A + (size_t)row0 * D_MODEL + kc*32 + 64), 0, 1);
    v16bf a0 = load_a_frag(A, D_MODEL, row0,      kc*32, lane);
    v16bf a1 = load_a_frag(A, D_MODEL, row0 + 16, kc*32, lane);
#pragma unroll
    for (int j = 0; j < 4; ++j) {
      v16bf b = load_b_packed(wp, D_MODEL, nt4*4 + j, kc, lane);
      acc[0][j] = wmma_bf16(a0, b, acc[0][j]);
      acc[1][j] = wmma_bf16(a1, b, acc[1][j]);
    }
  }

#pragma unroll
  for (int j = 0; j < 4; ++j) {
    const int col  = (nt4*4 + j)*16 + lrow;
    const float bv = bias[col];
#pragma unroll
    for (int i = 0; i < 2; ++i)
#pragma unroll
      for (int r = 0; r < 8; ++r) {
        const int m = mt2*32 + i*16 + r + 8*half;
        out[(size_t)m * D_MODEL + col] = acc[i][j][r] + bv;
      }
  }
}

// ---------------------------------------------------------------------------
// Launch
// ---------------------------------------------------------------------------
extern "C" void kernel_launch(void* const* d_in, const int* in_sizes, int n_in,
                              void* d_out, int out_size, void* d_ws, size_t ws_size,
                              hipStream_t stream) {
  const float* x     = (const float*)d_in[0];
  const int*   mask  = (const int*)  d_in[1];
  const float* Wqkv  = (const float*)d_in[2];
  const float* bqkv  = (const float*)d_in[3];
  const float* Wproj = (const float*)d_in[4];
  const float* bproj = (const float*)d_in[5];
  float* out = (float*)d_out;

  // workspace layout (bf16), ~40 MB total
  char* ws = (char*)d_ws;
  size_t off = 0;
  bf16* xb     = (bf16*)(ws + off); off += (size_t)M_TOK * D_MODEL * sizeof(bf16);
  bf16* wqkvp  = (bf16*)(ws + off); off += (size_t)D_MODEL * N_QKV * sizeof(bf16);
  bf16* wprojp = (bf16*)(ws + off); off += (size_t)D_MODEL * D_MODEL * sizeof(bf16);
  bf16* qb     = (bf16*)(ws + off); off += (size_t)M_TOK * D_MODEL * sizeof(bf16);
  bf16* kb     = (bf16*)(ws + off); off += (size_t)M_TOK * D_MODEL * sizeof(bf16);
  bf16* vt     = (bf16*)(ws + off); off += (size_t)M_TOK * D_MODEL * sizeof(bf16);
  bf16* attnb  = xb;  // xb is dead after the QKV GEMM; reuse for attn output

  cvt_f32_bf16<<<2048, 256, 0, stream>>>(x, xb, M_TOK * D_MODEL);
  pack_w_frag <<<2048, 256, 0, stream>>>(Wqkv,  wqkvp,  D_MODEL, N_QKV);
  pack_w_frag <<<1024, 256, 0, stream>>>(Wproj, wprojp, D_MODEL, D_MODEL);

  // QKV GEMM: (4096/32)*(3072/64) = 6144 waves -> 768 blocks
  qkv_gemm<<<(M_TOK/32) * (N_QKV/64) / 8, 256, 0, stream>>>(xb, wqkvp, bqkv, qb, kb, vt);

  // Attention: B*H*(S/32) = 2048 waves -> 256 blocks
  attn_kernel<<<BATCH * N_HEADS * (SEQ/32) / 8, 256, 0, stream>>>(qb, kb, vt, mask, attnb);

  // Projection: (4096/32)*(1024/64) = 2048 waves -> 256 blocks
  proj_gemm<<<(M_TOK/32) * (D_MODEL/64) / 8, 256, 0, stream>>>(attnb, wprojp, bproj, out);
}
